// TemporalAttention_67851893342429
// MI455X (gfx1250) — matrix operand order repaired and checked by reading, not compiled, not run on hardware
//
#include <hip/hip_runtime.h>
#include <hip/hip_bf16.h>
#include <math.h>
#include <stdint.h>

#define B_  4
#define L_  2048
#define D_  1024
#define H_  16
#define DKc 64

// LDS weight-slab layout: 128 rows x 1024 bf16, padded 8 elements (16B) after
// element 512 of each row => row stride 1040 elements (2080B). Lane-to-lane
// bank stride = 520 DW % 64 = 8 banks; all fragment loads stay 16B-aligned.
#define WROWE 1040
#define WSMEM (128 * WROWE * 2)

typedef __attribute__((ext_vector_type(16))) __bf16 v16bf;
typedef __attribute__((ext_vector_type(8)))  __bf16 v8bf;
typedef __attribute__((ext_vector_type(4)))  __bf16 v4bf;
typedef __attribute__((ext_vector_type(8)))  float  v8f;
typedef __attribute__((ext_vector_type(4)))  float  v4f;
typedef __attribute__((ext_vector_type(4)))  unsigned int v4u;
typedef __attribute__((ext_vector_type(8)))  int v8i;
typedef __attribute__((ext_vector_type(4)))  int v4i;

__device__ __forceinline__ v8f wmma_bf16(v16bf a, v16bf b, v8f c) {
  return __builtin_amdgcn_wmma_f32_16x16x32_bf16(false, a, false, b, (short)0, c,
                                                 false, false);
}

__device__ __forceinline__ v8f zero8() {
  v8f z = {0.f, 0.f, 0.f, 0.f, 0.f, 0.f, 0.f, 0.f};
  return z;
}

// ---- fragment loaders -------------------------------------------------------
// A matrix 16x32 bf16 (M x K): lane m (0..15) holds row M=m, K 0..7 & 16..23;
// lane m+16 holds K 8..15 & 24..31.
__device__ __forceinline__ v16bf fragA_bf16(const __bf16* __restrict__ p) {
  const int o = ((threadIdx.x >> 4) & 1) << 3;
  v8bf lo = *(const v8bf*)(p + o);
  v8bf hi = *(const v8bf*)(p + o + 16);
  v16bf r;
#pragma unroll
  for (int i = 0; i < 8; ++i) { r[i] = lo[i]; r[8 + i] = hi[i]; }
  return r;
}

// B matrix 32x16 bf16 (K x N): lane n holds column n; lanes 0-15: K 0..7 & 16..23,
// lanes 16-31: K 8..15 & 24..31.  `p` = memory row holding column n (K contiguous).
__device__ __forceinline__ v16bf fragB_bf16(const __bf16* __restrict__ p) {
  const int o = ((threadIdx.x >> 4) & 1) << 3;
  v8bf lo = *(const v8bf*)(p + o);
  v8bf hi = *(const v8bf*)(p + o + 16);
  v16bf r;
#pragma unroll
  for (int i = 0; i < 8; ++i) { r[i] = lo[i]; r[8 + i] = hi[i]; }
  return r;
}

// B fragment from the padded LDS weight slab; `base` = LDS row base.
__device__ __forceinline__ v16bf fragB_lds(const __bf16* base, int kd) {
  const int sel = (threadIdx.x >> 4) & 1;
  int e = kd + (sel << 3);
  int e2 = e + 16;
  e += (e >> 9) << 3;  // skip the 8-element pad after element 512
  e2 += (e2 >> 9) << 3;
  v8bf lo = *(const v8bf*)(base + e);
  v8bf hi = *(const v8bf*)(base + e2);
  v16bf r;
#pragma unroll
  for (int i = 0; i < 8; ++i) { r[i] = lo[i]; r[8 + i] = hi[i]; }
  return r;
}

// A fragment from f32 source (only used for the attn matrix in k_ctx).
__device__ __forceinline__ v16bf fragA_f32(const float* __restrict__ p) {
  const int o = ((threadIdx.x >> 4) & 1) << 3;
  v4f a0 = *(const v4f*)(p + o);
  v4f a1 = *(const v4f*)(p + o + 4);
  v4f a2 = *(const v4f*)(p + o + 16);
  v4f a3 = *(const v4f*)(p + o + 20);
  v16bf r;
#pragma unroll
  for (int i = 0; i < 4; ++i) {
    r[i] = (__bf16)a0[i];
    r[4 + i] = (__bf16)a1[i];
    r[8 + i] = (__bf16)a2[i];
    r[12 + i] = (__bf16)a3[i];
  }
  return r;
}

// ---- TDM staging of a 128 x D_ bf16 weight slab into padded LDS ------------
__device__ __forceinline__ void stage_wslab(const __bf16* __restrict__ Wslab,
                                            __bf16* lds) {
#if __has_builtin(__builtin_amdgcn_tensor_load_to_lds) && \
    __has_builtin(__builtin_amdgcn_s_wait_tensorcnt)
  if (threadIdx.x < 32) {  // wave 0 issues one TDM descriptor for the block
    const uint64_t ga = (uint64_t)(const void*)Wslab;
    // D# group 0: count=1, lds_addr=0 (dynamic LDS base), global addr, type=2
    v4u g0;
    g0[0] = 1u;
    g0[1] = 0u;
    g0[2] = (unsigned)(ga & 0xffffffffu);
    g0[3] = (unsigned)((ga >> 32) & 0x01ffffffu) | (2u << 30);
    // D# group 1: data_size=2B; pad_enable, interval=256DW(code7),
    // amount=4DW(code3); tensor_dim0=D_, tensor_dim1=128; tile 1024x128;
    // dim0 stride = D_.
    v8i g1;
    g1[0] = (int)((1u << 16) | (1u << 20) | (7u << 22) | (3u << 25));
    g1[1] = (int)(((unsigned)D_ & 0xffffu) << 16);          // tdim0[15:0]
    g1[2] = (int)((((unsigned)D_ >> 16) & 0xffffu) |        // tdim0[31:16]
                  ((128u & 0xffffu) << 16));                // tdim1[15:0]
    g1[3] = (int)(((unsigned)D_ & 0xffffu) << 16);          // tile_dim0=D_
    g1[4] = 128;                                            // tile_dim1=128
    g1[5] = D_;                                             // dim0 stride lo32
    g1[6] = 0;
    g1[7] = 0;
    v4i g2 = {0, 0, 0, 0};
    v4i g3 = {0, 0, 0, 0};
    v8i g4 = {0, 0, 0, 0, 0, 0, 0, 0};
    // 6-arg toolchain variant: (v4u, v8i, v4i, v4i, v8i, i32 cpol)
    __builtin_amdgcn_tensor_load_to_lds(g0, g1, g2, g3, g4, 0);
    __builtin_amdgcn_s_wait_tensorcnt(0);
  }
  __syncthreads();
#else
  // cooperative fallback with identical padded layout
  for (int idx = threadIdx.x; idx < 128 * (D_ / 16); idx += 256) {
    const int r = idx / (D_ / 16);
    const int e = (idx % (D_ / 16)) * 16;
    const v8bf* src = (const v8bf*)(Wslab + (size_t)r * D_ + e);
    const int eo = e + ((e >> 9) << 3);
    v8bf* dst = (v8bf*)(lds + (size_t)r * WROWE + eo);
    dst[0] = src[0];
    dst[1] = src[1];
  }
  __syncthreads();
#endif
}

// ---- K-1: elementwise f32 -> bf16 (x and weights, once) --------------------
__global__ __launch_bounds__(256) void k_cvt(const float* __restrict__ src,
                                             __bf16* __restrict__ dst, int n4) {
  const int i = blockIdx.x * 256 + threadIdx.x;
  if (i >= n4) return;
  v4f f = *(const v4f*)(src + (size_t)i * 4);
  v4bf o;
#pragma unroll
  for (int j = 0; j < 4; ++j) o[j] = (__bf16)f[j];
  *(v4bf*)(dst + (size_t)i * 4) = o;
}

// ---- K0: cumulative sum of time_deltas per batch ---------------------------
__global__ void k_cumsum(const float* __restrict__ td, float* __restrict__ cum) {
  int b = threadIdx.x;
  if (b >= B_) return;
  float s = 0.f;
  const float* t = td + b * L_;
  float* c = cum + b * L_;
  for (int i = 0; i < L_; ++i) { s += t[i]; c[i] = s; }
}

// ---- K1: QKV projections, TDM-staged weights, WMMA -------------------------
// grid (BL/64, D/128, 3), block 256, dyn-LDS WSMEM. blockIdx.z: 0=Q,1=K,2=V
__global__ __launch_bounds__(256) void k_qkv(
    const __bf16* __restrict__ xb,
    const __bf16* __restrict__ Wqb, const __bf16* __restrict__ Wkb,
    const __bf16* __restrict__ Wvb,
    const float* __restrict__ bq, const float* __restrict__ bk,
    const float* __restrict__ bv,
    __bf16* __restrict__ Qbf, __bf16* __restrict__ Kbf, __bf16* __restrict__ Vt) {
  extern __shared__ __bf16 Wlds[];
  const int lane = threadIdx.x & 31;
  const int wave = threadIdx.x >> 5;
  const int which = blockIdx.z;
  const int rt = wave & 3;
  const int cg = wave >> 2;
  const int grow0 = blockIdx.x * 64 + rt * 16;
  const int col0 = blockIdx.y * 128;
  const int m = lane & 15;

  const __bf16* W = (which == 0) ? Wqb : (which == 1) ? Wkb : Wvb;
  const float* bias = (which == 0) ? bq : (which == 1) ? bk : bv;

  stage_wslab(W + (size_t)col0 * D_, Wlds);

  const __bf16* arow = xb + (size_t)(grow0 + m) * D_;
  const __bf16* lrow[4];
#pragma unroll
  for (int nt = 0; nt < 4; ++nt)
    lrow[nt] = Wlds + (size_t)(cg * 64 + nt * 16 + m) * WROWE;

  v8f acc[4];
#pragma unroll
  for (int nt = 0; nt < 4; ++nt) acc[nt] = zero8();

  for (int kd = 0; kd < D_; kd += 32) {
    v16bf a = fragA_bf16(arow + kd);
    v16bf bf[4];
#pragma unroll
    for (int nt = 0; nt < 4; ++nt) bf[nt] = fragB_lds(lrow[nt], kd);
#pragma unroll
    for (int nt = 0; nt < 4; ++nt) acc[nt] = wmma_bf16(a, bf[nt], acc[nt]);
  }

#pragma unroll
  for (int nt = 0; nt < 4; ++nt) {
    const int col = col0 + cg * 64 + nt * 16 + m;
    const float bvv = bias[col];
    const int h = col >> 6, dk = col & 63;
#pragma unroll
    for (int r = 0; r < 8; ++r) {
      const int grow = grow0 + r + (((lane >> 4) & 1) << 3);
      const int b = grow >> 11, l = grow & (L_ - 1);
      const __bf16 o = (__bf16)(acc[nt][r] + bvv);
      if (which == 2)
        Vt[((size_t)(b * H_ + h) * DKc + dk) * L_ + l] = o;
      else if (which == 0)
        Qbf[((size_t)(b * H_ + h) * L_ + l) * DKc + dk] = o;
      else
        Kbf[((size_t)(b * H_ + h) * L_ + l) * DKc + dk] = o;
    }
  }
}

// ---- K2: scores = QK^T/8 - softplus(bias), causal+pad mask -----------------
// grid (L/64, B*H), block 128 (4 waves, one 16-row q-tile each)
__global__ __launch_bounds__(128) void k_scores(
    const __bf16* __restrict__ Qbf, const __bf16* __restrict__ Kbf,
    const float* __restrict__ cum, const int* __restrict__ mask,
    const float* __restrict__ tb_w, const float* __restrict__ tb_b,
    float* __restrict__ attn) {
  const int lane = threadIdx.x & 31;
  const int wave = threadIdx.x >> 5;
  const int bh = blockIdx.y;
  const int b = bh >> 4, h = bh & (H_ - 1);
  const int q0 = blockIdx.x * 64 + wave * 16;
  const int m = lane & 15;
  const float tw = tb_w[h], tbb = tb_b[h];

  const __bf16* qrow = Qbf + ((size_t)bh * L_ + q0 + m) * DKc;
  const v16bf aq0 = fragA_bf16(qrow);
  const v16bf aq1 = fragA_bf16(qrow + 32);

  float cumq[8];
#pragma unroll
  for (int r = 0; r < 8; ++r)
    cumq[r] = cum[b * L_ + q0 + r + (((lane >> 4) & 1) << 3)];

  const int ktmax = q0 >> 4;
  for (int kt = 0; kt <= ktmax; ++kt) {
    const int k0 = kt << 4;
    const __bf16* krow = Kbf + ((size_t)bh * L_ + k0 + m) * DKc;
    v16bf b0 = fragB_bf16(krow);
    v16bf b1 = fragB_bf16(krow + 32);
    v8f acc = zero8();
    acc = wmma_bf16(aq0, b0, acc);
    acc = wmma_bf16(aq1, b1, acc);

    const int kcol = k0 + m;
    const float cumk = cum[b * L_ + kcol];
    const int mk = mask[b * L_ + kcol];
#pragma unroll
    for (int r = 0; r < 8; ++r) {
      const int qrowi = q0 + r + (((lane >> 4) & 1) << 3);
      float s = acc[r] * 0.125f;  // 1/sqrt(64)
      const float zv = fabsf(cumq[r] - cumk) * tw + tbb;
      const float sp = fmaxf(zv, 0.f) + log1pf(__expf(-fabsf(zv)));
      s -= sp;
      if (kcol > qrowi || mk == 0) s = -1e30f;
      attn[((size_t)bh * L_ + qrowi) * L_ + kcol] = s;
    }
  }
}

// ---- K3: row softmax in place; zeros the masked (upper) triangle -----------
__global__ __launch_bounds__(256) void k_softmax(float* __restrict__ attn) {
  const int row = blockIdx.x;
  const int q = row & (L_ - 1);
  const int valid = q + 1;
  float* p = attn + (size_t)row * L_;
  const int tid = threadIdx.x;
  __shared__ float red[256];

  float v[8];
#pragma unroll
  for (int i = 0; i < 8; ++i) {
    const int g = tid * 8 + i;
    v[i] = (g < valid) ? p[g] : -1e30f;
  }
  float mx = -1e30f;
#pragma unroll
  for (int i = 0; i < 8; ++i) mx = fmaxf(mx, v[i]);
  red[tid] = mx;
  __syncthreads();
  for (int s = 128; s > 0; s >>= 1) {
    if (tid < s) red[tid] = fmaxf(red[tid], red[tid + s]);
    __syncthreads();
  }
  mx = red[0];
  __syncthreads();

  float sum = 0.f;
#pragma unroll
  for (int i = 0; i < 8; ++i) {
    const int g = tid * 8 + i;
    v[i] = (g < valid) ? __expf(v[i] - mx) : 0.f;
    sum += v[i];
  }
  red[tid] = sum;
  __syncthreads();
  for (int s = 128; s > 0; s >>= 1) {
    if (tid < s) red[tid] += red[tid + s];
    __syncthreads();
  }
  const float inv = 1.f / red[0];
#pragma unroll
  for (int i = 0; i < 8; ++i) p[tid * 8 + i] = v[i] * inv;
}

// ---- K4: ctx = attn @ V (V stored transposed), causal-limited --------------
// grid (L/64, B*H), block 128
__global__ __launch_bounds__(128) void k_ctx(
    const float* __restrict__ attn, const __bf16* __restrict__ Vt,
    __bf16* __restrict__ ctxbf) {
  const int lane = threadIdx.x & 31;
  const int wave = threadIdx.x >> 5;
  const int bh = blockIdx.y;
  const int b = bh >> 4, h = bh & (H_ - 1);
  const int q0 = blockIdx.x * 64 + wave * 16;
  const int m = lane & 15;

  const float* arow = attn + ((size_t)bh * L_ + q0 + m) * L_;
  const __bf16* vrow[4];
#pragma unroll
  for (int nt = 0; nt < 4; ++nt)
    vrow[nt] = Vt + ((size_t)bh * DKc + nt * 16 + m) * L_;

  v8f acc[4];
#pragma unroll
  for (int nt = 0; nt < 4; ++nt) acc[nt] = zero8();

  const int kend = q0 + 16;  // attn is zero beyond the diagonal
  for (int k0 = 0; k0 < kend; k0 += 32) {
    v16bf bf[4];
#pragma unroll
    for (int nt = 0; nt < 4; ++nt) bf[nt] = fragB_bf16(vrow[nt] + k0);
    v16bf a = fragA_f32(arow + k0);
#pragma unroll
    for (int nt = 0; nt < 4; ++nt) acc[nt] = wmma_bf16(a, bf[nt], acc[nt]);
  }

#pragma unroll
  for (int nt = 0; nt < 4; ++nt) {
    const int col = h * DKc + nt * 16 + m;
#pragma unroll
    for (int r = 0; r < 8; ++r) {
      const int l = q0 + r + (((lane >> 4) & 1) << 3);
      ctxbf[(size_t)(b * L_ + l) * D_ + col] = (__bf16)acc[nt][r];
    }
  }
}

// ---- K5: out = ctx @ Wo.T + bo, TDM-staged Wo ------------------------------
// grid (BL/64, D/128), block 256, dyn-LDS WSMEM
__global__ __launch_bounds__(256) void k_outproj(
    const __bf16* __restrict__ ctxbf, const __bf16* __restrict__ Wob,
    const float* __restrict__ bo, float* __restrict__ out) {
  extern __shared__ __bf16 Wlds[];
  const int lane = threadIdx.x & 31;
  const int wave = threadIdx.x >> 5;
  const int rt = wave & 3;
  const int cg = wave >> 2;
  const int grow0 = blockIdx.x * 64 + rt * 16;
  const int col0 = blockIdx.y * 128;
  const int m = lane & 15;

  stage_wslab(Wob + (size_t)col0 * D_, Wlds);

  const __bf16* arow = ctxbf + (size_t)(grow0 + m) * D_;
  const __bf16* lrow[4];
#pragma unroll
  for (int nt = 0; nt < 4; ++nt)
    lrow[nt] = Wlds + (size_t)(cg * 64 + nt * 16 + m) * WROWE;

  v8f acc[4];
#pragma unroll
  for (int nt = 0; nt < 4; ++nt) acc[nt] = zero8();

  for (int kd = 0; kd < D_; kd += 32) {
    v16bf a = fragA_bf16(arow + kd);
    v16bf bf[4];
#pragma unroll
    for (int nt = 0; nt < 4; ++nt) bf[nt] = fragB_lds(lrow[nt], kd);
#pragma unroll
    for (int nt = 0; nt < 4; ++nt) acc[nt] = wmma_bf16(a, bf[nt], acc[nt]);
  }

#pragma unroll
  for (int nt = 0; nt < 4; ++nt) {
    const int col = col0 + cg * 64 + nt * 16 + m;
    const float bvv = bo[col];
#pragma unroll
    for (int r = 0; r < 8; ++r) {
      const int grow = grow0 + r + (((lane >> 4) & 1) << 3);
      out[(size_t)grow * D_ + col] = acc[nt][r] + bvv;
    }
  }
}

// ---- launcher ---------------------------------------------------------------
extern "C" void kernel_launch(void* const* d_in, const int* in_sizes, int n_in,
                              void* d_out, int out_size, void* d_ws,
                              size_t ws_size, hipStream_t stream) {
  const float* x   = (const float*)d_in[0];
  const int*   msk = (const int*)d_in[1];
  const float* td  = (const float*)d_in[2];
  const float* Wq  = (const float*)d_in[3];
  const float* bq  = (const float*)d_in[4];
  const float* Wk  = (const float*)d_in[5];
  const float* bk  = (const float*)d_in[6];
  const float* Wv  = (const float*)d_in[7];
  const float* bv  = (const float*)d_in[8];
  const float* Wo  = (const float*)d_in[9];
  const float* bo  = (const float*)d_in[10];
  const float* tbw = (const float*)d_in[11];
  const float* tbb = (const float*)d_in[12];

  float* out  = (float*)d_out;                         // [B, L, D]
  float* attn = (float*)d_out + (size_t)B_ * L_ * D_;  // [B, H, L, L]

  const size_t NBHLD = (size_t)B_ * H_ * L_ * DKc;     // 8Mi elements
  const size_t NW    = (size_t)D_ * D_;                // 1Mi elements
  char* ws = (char*)d_ws;
  size_t off = 0;
  __bf16* Qbf = (__bf16*)(ws + off); off += NBHLD * 2;
  __bf16* Kbf = (__bf16*)(ws + off); off += NBHLD * 2;
  __bf16* Vt  = (__bf16*)(ws + off); off += NBHLD * 2;
  __bf16* Ctx = (__bf16*)(ws + off); off += NBHLD * 2;
  __bf16* Xb  = (__bf16*)(ws + off); off += NBHLD * 2;
  __bf16* Wqb = (__bf16*)(ws + off); off += NW * 2;
  __bf16* Wkb = (__bf16*)(ws + off); off += NW * 2;
  __bf16* Wvb = (__bf16*)(ws + off); off += NW * 2;
  __bf16* Wob = (__bf16*)(ws + off); off += NW * 2;
  float*  cum = (float*)(ws + off);  off += (size_t)B_ * L_ * 4;

  (void)hipFuncSetAttribute((const void*)k_qkv,
                            hipFuncAttributeMaxDynamicSharedMemorySize, WSMEM);
  (void)hipFuncSetAttribute((const void*)k_outproj,
                            hipFuncAttributeMaxDynamicSharedMemorySize, WSMEM);

  // one-time f32 -> bf16 mirrors
  k_cvt<<<dim3((int)(NBHLD / 4 / 256)), 256, 0, stream>>>(x, Xb, (int)(NBHLD / 4));
  k_cvt<<<dim3((int)(NW / 4 / 256)), 256, 0, stream>>>(Wq, Wqb, (int)(NW / 4));
  k_cvt<<<dim3((int)(NW / 4 / 256)), 256, 0, stream>>>(Wk, Wkb, (int)(NW / 4));
  k_cvt<<<dim3((int)(NW / 4 / 256)), 256, 0, stream>>>(Wv, Wvb, (int)(NW / 4));
  k_cvt<<<dim3((int)(NW / 4 / 256)), 256, 0, stream>>>(Wo, Wob, (int)(NW / 4));

  k_cumsum<<<1, 32, 0, stream>>>(td, cum);

  k_qkv<<<dim3((B_ * L_) / 64, D_ / 128, 3), 256, WSMEM, stream>>>(
      Xb, Wqb, Wkb, Wvb, bq, bk, bv, Qbf, Kbf, Vt);

  k_scores<<<dim3(L_ / 64, B_ * H_), 128, 0, stream>>>(
      Qbf, Kbf, cum, msk, tbw, tbb, attn);

  k_softmax<<<dim3(B_ * H_ * L_), 256, 0, stream>>>(attn);

  k_ctx<<<dim3(L_ / 64, B_ * H_), 128, 0, stream>>>(attn, Vt, Ctx);

  k_outproj<<<dim3((B_ * L_) / 64, D_ / 128), 256, WSMEM, stream>>>(
      Ctx, Wob, bo, out);
}
